// SOM_68547678044305
// MI455X (gfx1250) — matrix operand order repaired and checked by reading, not compiled, not run on hardware
//
#include <hip/hip_runtime.h>

typedef __attribute__((ext_vector_type(16))) _Float16 v16h;
typedef __attribute__((ext_vector_type(8)))  float    v8f;

#define SOM_B    512
#define SOM_MN   1600
#define SOM_DIM  256
#define GRID_N   40

// ---------------------------------------------------------------------------
// Kernel 1: BMU search.  argmin_n ( w2[n] - 2 * x[b] . w[n] )   (x2 dropped)
// grid = 32 blocks (16 samples each), block = 128 threads (4 waves).
// GEMM1 tile: 16 samples x 16 neurons, K=256 in 8 steps of v_wmma f16 K=32.
// ---------------------------------------------------------------------------
__global__ __launch_bounds__(128)
void som_bmu_kernel(const float* __restrict__ X,
                    const float* __restrict__ W,
                    int* __restrict__ bmu)
{
    __shared__ _Float16 sX[16][SOM_DIM];   // 8 KB sample tile (f16)
    __shared__ float    sVal[16][64];      // per-row argmin reduction
    __shared__ int      sIdx[16][64];

    const int tid   = threadIdx.x;
    const int lane  = tid & 31;
    const int wid   = tid >> 5;
    const int s0    = blockIdx.x * 16;
    const int nloc  = lane & 15;     // A-row M / B-col N within tile
    const int khalf = lane >> 4;     // 0 or 1: which K half this lane owns

    // cooperative load + f32 -> f16 convert of 16x256 sample tile
    for (int i = tid; i < 16 * SOM_DIM; i += 128) {
        sX[i >> 8][i & 255] = (_Float16)X[(s0 + (i >> 8)) * SOM_DIM + (i & 255)];
    }
    __syncthreads();

    // Build the eight 16x32 f16 A-fragments once per wave (A layout:
    // K = 16*(lane/16) + 2*p + h, matching B)
    v16h a[8];
#pragma unroll
    for (int ks = 0; ks < 8; ++ks) {
#pragma unroll
        for (int p = 0; p < 8; ++p) {
            const int k = 32 * ks + 16 * khalf + 2 * p;
            a[ks][2 * p]     = sX[nloc][k];
            a[ks][2 * p + 1] = sX[nloc][k + 1];
        }
    }

    float bestv[8];
    int   besti[8];
#pragma unroll
    for (int r = 0; r < 8; ++r) { bestv[r] = 3.0e38f; besti[r] = 0x7FFFFFFF; }

    // each wave sweeps neuron tiles wid, wid+4, ... (25 tiles)
    for (int t = wid; t < SOM_MN / 16; t += 4) {
        const float* wrow = W + (t * 16 + nloc) * SOM_DIM + 16 * khalf;
        // CDNA5 speculative prefetch of next tile's slice (global_prefetch_b8)
        __builtin_prefetch(wrow + 4 * 16 * SOM_DIM, 0, 1);

        v8f   acc = {};
        float w2p = 0.0f;     // fp32 partial sum of w^2 over this lane's K set
#pragma unroll
        for (int ks = 0; ks < 8; ++ks) {
            v16h b;           // B layout: N = lane%16, K = 16*(lane/16) + e
#pragma unroll
            for (int e = 0; e < 16; ++e) {
                const float v = wrow[32 * ks + e];
                b[e] = (_Float16)v;
                w2p += v * v;
            }
            acc = __builtin_amdgcn_wmma_f32_16x16x32_f16(
                false, a[ks], false, b, (short)0, acc, false, false);
        }
        const float w2   = w2p + __shfl_xor(w2p, 16, 32);  // lanes l / l^16
        const int   nidx = t * 16 + nloc;
#pragma unroll
        for (int r = 0; r < 8; ++r) {
            const float score = w2 - 2.0f * acc[r];        // d2 minus const x2
            if (score < bestv[r] || (score == bestv[r] && nidx < besti[r])) {
                bestv[r] = score;
                besti[r] = nidx;
            }
        }
    }

    // C layout: row M = r + 8*khalf, col = this lane's neuron slot
#pragma unroll
    for (int r = 0; r < 8; ++r) {
        const int row = r + 8 * khalf;
        sVal[row][wid * 16 + nloc] = bestv[r];
        sIdx[row][wid * 16 + nloc] = besti[r];
    }
    __syncthreads();

    if (tid < 16) {
        float bv = 3.0e38f;
        int   bi = 0x7FFFFFFF;
        for (int c = 0; c < 64; ++c) {
            const float v = sVal[tid][c];
            const int  ix = sIdx[tid][c];
            if (v < bv || (v == bv && ix < bi)) { bv = v; bi = ix; }
        }
        bmu[s0 + tid] = bi;
    }
}

// ---------------------------------------------------------------------------
// Kernel 2: fused neighborhood + update.
//   out[m,d] = W[m,d] * (1 - meanlr[m]) + (1/B) * sum_b lr[b,m] * X[b,d]
// grid = 100 blocks (16 neurons each), block = 256 threads (8 waves).
// GEMM2: A = lr^T (generated in-register), B = X, K = 512 in 16 steps.
// ---------------------------------------------------------------------------
__global__ __launch_bounds__(256)
void som_update_kernel(const float* __restrict__ X,
                       const float* __restrict__ W,
                       const int* __restrict__ epoch_p,
                       const int* __restrict__ bmu,
                       float* __restrict__ out)
{
    __shared__ int   sBmu[SOM_B];
    __shared__ float sSum[16];

    const int tid   = threadIdx.x;
    const int lane  = tid & 31;
    const int wid   = tid >> 5;
    const int nloc  = lane & 15;
    const int khalf = lane >> 4;

    for (int i = tid; i < SOM_B; i += 256) sBmu[i] = bmu[i];
    __syncthreads();

    const float decay  = 1.0f - (float)epoch_p[0] * 0.01f;
    const float alpha  = 0.3f * decay;
    const float sigma  = 20.0f * decay;
    const float inv_s2 = 1.0f / (sigma * sigma);

    const int mtile = blockIdx.x;
    const int m     = mtile * 16 + nloc;          // A-row neuron (lane-const)
    const int mi    = m / GRID_N;
    const int mj    = m % GRID_N;

    const int d0 = (wid * 2 + 0) * 16 + nloc;     // two 16-wide dim tiles
    const int d1 = (wid * 2 + 1) * 16 + nloc;

    v8f   acc0 = {}, acc1 = {};
    float lrsum = 0.0f;                           // fp32 row-sum of lr

    for (int ks = 0; ks < 16; ++ks) {
        v16h afrag;                               // lr^T fragment, built in-reg
#pragma unroll
        for (int p = 0; p < 8; ++p) {
#pragma unroll
            for (int h = 0; h < 2; ++h) {
                const int   b  = 32 * ks + 16 * khalf
                               + 2 * p + h;
                const int   bi = sBmu[b];
                const int   di = mi - bi / GRID_N;
                const int   dj = mj - bi % GRID_N;
                const float g2 = (float)(di * di + dj * dj);
                const float lr = alpha * __expf(-g2 * inv_s2);
                lrsum += lr;
                afrag[2 * p + h] = (_Float16)lr;
            }
        }
        v16h b0, b1;                              // B = X tile, N = dim
#pragma unroll
        for (int e = 0; e < 16; ++e) {
            const int brow = 32 * ks + 16 * khalf + e;
            b0[e] = (_Float16)X[brow * SOM_DIM + d0];
            b1[e] = (_Float16)X[brow * SOM_DIM + d1];
        }
        acc0 = __builtin_amdgcn_wmma_f32_16x16x32_f16(
            false, afrag, false, b0, (short)0, acc0, false, false);
        acc1 = __builtin_amdgcn_wmma_f32_16x16x32_f16(
            false, afrag, false, b1, (short)0, acc1, false, false);
    }

    // lanes l and l^16 cover complementary batch halves for the same row
    const float sum = lrsum + __shfl_xor(lrsum, 16, 32);
    if (wid == 0 && lane < 16) sSum[lane] = sum;  // all waves agree; wave0 writes
    __syncthreads();

    const float invB = 1.0f / (float)SOM_B;
#pragma unroll
    for (int r = 0; r < 8; ++r) {
        const int   mrow  = r + 8 * khalf;        // C layout: M = r + 8*(l/16)
        const int   mm    = mtile * 16 + mrow;
        const float scale = 1.0f - sSum[mrow] * invB;
        out[mm * SOM_DIM + d0] = W[mm * SOM_DIM + d0] * scale + acc0[r] * invB;
        out[mm * SOM_DIM + d1] = W[mm * SOM_DIM + d1] * scale + acc1[r] * invB;
    }
}

// ---------------------------------------------------------------------------
extern "C" void kernel_launch(void* const* d_in, const int* in_sizes, int n_in,
                              void* d_out, int out_size, void* d_ws, size_t ws_size,
                              hipStream_t stream)
{
    (void)in_sizes; (void)n_in; (void)out_size; (void)ws_size;
    const float* X     = (const float*)d_in[0];   // [512, 256] f32
    const float* W     = (const float*)d_in[1];   // [1600, 256] f32
    const int*   epoch = (const int*)d_in[2];     // [1] i32
    float*       out   = (float*)d_out;           // [1600, 256] f32
    int*         bmu   = (int*)d_ws;              // 512 ints of scratch

    som_bmu_kernel<<<SOM_B / 16, 128, 0, stream>>>(X, W, bmu);
    som_update_kernel<<<SOM_MN / 16, 256, 0, stream>>>(X, W, epoch, bmu, out);
}
